// Stage_137438953619
// MI455X (gfx1250) — compile-verified, matrix operand order checked by reading, not run on hardware
//
#include <hip/hip_runtime.h>
#include <hip/hip_bf16.h>

typedef __bf16 bf16t;
typedef __attribute__((ext_vector_type(16))) __bf16 v16bf;
typedef __attribute__((ext_vector_type(8)))  float  v8f;
typedef __attribute__((ext_vector_type(4)))  unsigned tdm_g0_t;
typedef __attribute__((ext_vector_type(8)))  unsigned tdm_g1_t;

#define TKN   24576   // D*H*W = 6*64*64 tokens
#define CDIM  192
#define NHEAD 6
#define HDIM  32

// ---------------------------------------------------------------- device utils

__device__ __forceinline__ v8f zero8() {
  v8f z; for (int i = 0; i < 8; i++) z[i] = 0.f; return z;
}

// 16-bit A/B fragment K-striping for V_WMMA_*_16X16X32 (ISA 7.12.2):
// lanes 0-15 hold K in {0..7, 16..23}, lanes 16-31 hold K in {8..15, 24..31},
// consecutive element pairs are consecutive K.
__device__ __forceinline__ int kmap16(int e, int lane) {
  int v = e >> 1;
  int base = (v < 4) ? (v << 1) : (16 + ((v - 4) << 1));
  return base + (e & 1) + ((lane >> 4) << 3);
}

__device__ __forceinline__ v8f wmma_bf16(v16bf a, v16bf b, v8f c) {
  return __builtin_amdgcn_wmma_f32_16x16x32_bf16(false, a, false, b, (short)0, c,
                                                 false, false);
}

__device__ __forceinline__ float wave_sum32(float v) {
  for (int o = 1; o < 32; o <<= 1) v += __shfl_xor(v, o);
  return v;
}

// LDS aperture maps flat addr[31:0] directly to the LDS offset (ISA 10.2).
__device__ __forceinline__ unsigned lds_off32(const void* p) {
  return (unsigned)(unsigned long long)p;
}

// CDNA5 async global->LDS copy, 16 bytes per active lane (ASYNCcnt tracked).
__device__ __forceinline__ void async_copy16(void* lds, const void* g) {
  unsigned loff = lds_off32(lds);
  asm volatile("global_load_async_to_lds_b128 %0, %1, off"
               :: "v"(loff), "v"(g) : "memory");
}
__device__ __forceinline__ void wait_async0() {
  asm volatile("s_wait_asynccnt 0x0" ::: "memory");
}

// CDNA5 Tensor Data Mover: load a 2D tile (rows x 32 elems of 2B) whose rows are
// row_stride elements apart in memory, packed contiguously into LDS.
__device__ __forceinline__ void tdm_load_2d(unsigned lds_addr, unsigned long long gaddr,
                                            unsigned rows, unsigned row_stride) {
  tdm_g0_t g0;
  g0[0] = 1u;                                                  // count=1 (valid user D#)
  g0[1] = lds_addr;                                            // lds_addr
  g0[2] = (unsigned)(gaddr & 0xFFFFFFFFu);                     // global_addr[31:0]
  g0[3] = (unsigned)((gaddr >> 32) & 0x01FFFFFFu) | (2u << 30);// global_addr[56:32] | type=2
  tdm_g1_t g1;
  g1[0] = 1u << 16;                        // workgroup_mask=0, data_size=1 (2 bytes)
  g1[1] = 32u << 16;                       // tensor_dim0[15:0]=32 (bits 63:48)
  g1[2] = (rows & 0xFFFFu) << 16;          // tensor_dim0 hi=0 | tensor_dim1[15:0]
  g1[3] = (rows >> 16) | (32u << 16);      // tensor_dim1 hi | tile_dim0=32
  g1[4] = rows & 0xFFFFu;                  // tile_dim1=rows, tile_dim2=0
  g1[5] = row_stride;                      // tensor_dim0_stride[31:0] (elem units)
  g1[6] = 0u;                              // stride hi16 | tensor_dim1_stride lo16
  g1[7] = 0u;
  asm volatile("tensor_load_to_lds %0, %1" :: "s"(g0), "s"(g1) : "memory");
}

// window-token -> spatial-token map (includes cyclic shift = roll by -s)
__device__ __forceinline__ int win_to_spatial(int wt, int mode, int s0, int s1, int s2) {
  if (mode == 2) return wt;
  if (mode == 0) {
    int win = wt >> 7, n = wt & 127;
    int pd = n >> 6, ph = (n >> 3) & 7, pw = n & 7;
    int wd = win >> 6, wh = (win >> 3) & 7, ww = win & 7;
    int d = wd * 2 + pd + s0; if (d >= 6) d -= 6;
    int h = (wh * 8 + ph + s1) & 63;
    int w = (ww * 8 + pw + s2) & 63;
    return (d * 64 + h) * 64 + w;
  } else {
    int win = wt / 384, n = wt - win * 384;
    int pd = n >> 6, ph = (n >> 3) & 7, pw = n & 7;
    int wh = win >> 3, ww = win & 7;
    return (pd * 64 + wh * 8 + ph) * 64 + ww * 8 + pw;
  }
}

// shift-region label for g1 shifted block (ws=(2,8,8), ss=(1,4,4), dims (6,64,64))
__device__ __forceinline__ int g1_label(int win, int p) {
  int pd = p >> 6, ph = (p >> 3) & 7, pw = p & 7;
  int wd = win >> 6, wh = (win >> 3) & 7, ww = win & 7;
  int d = wd * 2 + pd, h = wh * 8 + ph, w = ww * 8 + pw;
  int cd = (d < 4) ? 0 : (d < 5 ? 1 : 2);
  int ch = (h < 56) ? 0 : (h < 60 ? 1 : 2);
  int cw = (w < 56) ? 0 : (w < 60 ? 1 : 2);
  return (cd * 3 + ch) * 3 + cw;
}

__device__ __forceinline__ int rel_idx(int q, int k, int ws0, int ws1, int ws2) {
  int a = ws1 * ws2;
  int qd = q / a, qr = q - qd * a, qh = qr / ws2, qw = qr - qh * ws2;
  int kd = k / a, kr = k - kd * a, kh = kr / ws2, kw = kr - kh * ws2;
  return ((qd - kd + ws0 - 1) * (2 * ws1 - 1) + (qh - kh + ws1 - 1)) * (2 * ws2 - 1)
         + (qw - kw + ws2 - 1);
}

// ---------------------------------------------------------------- elementwise

__global__ void to_bf16_kernel(const float* __restrict__ s, bf16t* __restrict__ d, int n) {
  int i = blockIdx.x * 256 + threadIdx.x;
  if (i < n) d[i] = (bf16t)s[i];
}
__global__ void copy_kernel(float* __restrict__ d, const float* __restrict__ s, int n) {
  int i = blockIdx.x * 256 + threadIdx.x;
  if (i < n) d[i] = s[i];
}
__global__ void add_kernel(float* __restrict__ x, const float* __restrict__ y, int n) {
  int i = blockIdx.x * 256 + threadIdx.x;
  if (i < n) x[i] += y[i];
}
__global__ void add2_kernel(float* __restrict__ x, const float* __restrict__ a,
                            const float* __restrict__ b, int n) {
  int i = blockIdx.x * 256 + threadIdx.x;
  if (i < n) x[i] = a[i] + b[i];
}
__global__ void geglu_kernel(const float* __restrict__ h1, const float* __restrict__ h2,
                             bf16t* __restrict__ o, int n) {
  int i = blockIdx.x * 256 + threadIdx.x;
  if (i >= n) return;
  float x = h1[i];
  float g = 0.5f * x * (1.f + erff(x * 0.70710678118654752f));  // exact GELU
  o[i] = (bf16t)(g * h2[i]);
}
__global__ void transpose_in_kernel(const float* __restrict__ x, float* __restrict__ X, int n) {
  int i = blockIdx.x * 256 + threadIdx.x;
  if (i >= n) return;
  int t = i / CDIM, c = i - t * CDIM;
  X[i] = x[(size_t)c * TKN + t];   // (C,D,H,W) -> token-major
}
__global__ void transpose_out_kernel(const float* __restrict__ X, float* __restrict__ o, int n) {
  int i = blockIdx.x * 256 + threadIdx.x;
  if (i >= n) return;
  int t = i / CDIM, c = i - t * CDIM;
  o[(size_t)c * TKN + t] = X[i];
}
__global__ void unwindow_add_kernel(float* __restrict__ X, const float* __restrict__ P,
                                    int mode, int s0, int s1, int s2) {
  int i = blockIdx.x * 256 + threadIdx.x;
  if (i >= TKN * CDIM) return;
  int wt = i / CDIM, c = i - wt * CDIM;
  int t = win_to_spatial(wt, mode, s0, s1, s2);
  X[(size_t)t * CDIM + c] += P[i];
}
__global__ void pos_kernel(float* __restrict__ pos) {   // (128, 192) sine pos-enc
  int i = blockIdx.x * 256 + threadIdx.x;
  if (i >= 128 * CDIM) return;
  int n = i / CDIM, c = i - n * CDIM;
  int ih = (n >> 3) & 7, jw = n & 7;
  float denom = 8.0f + 1e-6f, s2pi = 6.283185307179586f;
  float ax = (c < 96) ? ((ih + 1) / denom * s2pi) : ((jw + 1) / denom * s2pi);
  int cc = (c < 96) ? c : c - 96;
  int p = cc >> 1;
  float dt = powf(10000.f, (2.f * p) / 96.f);
  float arg = ax / dt;
  pos[i] = (cc & 1) ? cosf(arg) : sinf(arg);
}

// ---------------------------------------------------------------- LayerNorm (+window +pos)

__global__ __launch_bounds__(256) void ln_window_kernel(
    const float* __restrict__ X, const float* __restrict__ gw, const float* __restrict__ gb,
    bf16t* __restrict__ Y, bf16t* __restrict__ Yp, const float* __restrict__ pos,
    int mode, int s0, int s1, int s2)
{
  int wave = threadIdx.x >> 5, lane = threadIdx.x & 31;
  int wt = blockIdx.x * 8 + wave;
  int t = win_to_spatial(wt, mode, s0, s1, s2);
  float v[6]; float s = 0.f;
  for (int i = 0; i < 6; i++) { v[i] = X[(size_t)t * CDIM + i * 32 + lane]; s += v[i]; }
  float mu = wave_sum32(s) * (1.f / 192.f);
  float q = 0.f;
  for (int i = 0; i < 6; i++) { float d = v[i] - mu; q += d * d; }
  float rs = rsqrtf(wave_sum32(q) * (1.f / 192.f) + 1e-5f);
  int n = (mode == 0) ? (wt & 127) : 0;
  for (int i = 0; i < 6; i++) {
    int c = i * 32 + lane;
    float y = (v[i] - mu) * rs * gw[c] + gb[c];
    Y[(size_t)wt * CDIM + c] = (bf16t)y;
    if (Yp) Yp[(size_t)wt * CDIM + c] = (bf16t)(y + pos[n * CDIM + c]);
  }
}

// ---------------------------------------------------------------- WMMA GEMM (M×K)×(K×N)+bias

__global__ __launch_bounds__(256) void gemm_bias_kernel(
    const bf16t* __restrict__ A, const bf16t* __restrict__ W, const float* __restrict__ bias,
    float* __restrict__ Of, bf16t* __restrict__ Ob, int M, int N, int K)
{
  __shared__ bf16t As[128 * 32];
  __shared__ bf16t Bs[64 * 32];       // transposed: [n][k]
  const int tid = threadIdx.x, lane = tid & 31, wave = tid >> 5;
  const int m0 = blockIdx.y * 128, n0 = blockIdx.x * 64;
  const int n16 = lane & 15;
  v8f acc[4] = { zero8(), zero8(), zero8(), zero8() };

  for (int k0 = 0; k0 < K; k0 += 32) {
    for (int i = tid; i < 512; i += 256) {             // 128x32 A tile: async -> LDS
      int row = i >> 2, kc = (i & 3) << 3;
      async_copy16(&As[row * 32 + kc], &A[(size_t)(m0 + row) * K + k0 + kc]);
    }
    {                                                  // 32x64 W tile, store transposed
      int kr = tid >> 3, nc = (tid & 7) << 3;
      union { uint4 u; bf16t h[8]; } tmp;
      tmp.u = *(const uint4*)(&W[(size_t)(k0 + kr) * N + n0 + nc]);
      for (int j = 0; j < 8; j++) Bs[(nc + j) * 32 + kr] = tmp.h[j];
    }
    wait_async0();
    __syncthreads();
    v16bf a;
    for (int e = 0; e < 16; e++) a[e] = As[(wave * 16 + n16) * 32 + kmap16(e, lane)];
    for (int nt = 0; nt < 4; nt++) {
      v16bf b;
      for (int e = 0; e < 16; e++) b[e] = Bs[(nt * 16 + n16) * 32 + kmap16(e, lane)];
      acc[nt] = wmma_bf16(a, b, acc[nt]);
    }
    __syncthreads();
  }
  for (int nt = 0; nt < 4; nt++)
    for (int r = 0; r < 8; r++) {
      int row = m0 + wave * 16 + r + ((lane >> 4) << 3);
      int col = n0 + nt * 16 + n16;
      float vv = acc[nt][r] + bias[col];
      if (Ob) Ob[(size_t)row * N + col] = (bf16t)vv;
      else    Of[(size_t)row * N + col] = vv;
    }
}

// ---------------------------------------------------------------- flash window attention

__global__ __launch_bounds__(256) void attn_kernel(
    const bf16t* __restrict__ QKV, const float* __restrict__ RPB, bf16t* __restrict__ Out,
    int nW, int Nwin, int Nq, int Nk, int qb, int kb, int row_off,
    int Cout, int col_off, int use_rpb, int use_mask,
    int ws0, int ws1, int ws2, float scale, int nQB)
{
  __shared__ bf16t Ks[384 * 32];
  __shared__ bf16t Vs[384 * 32];
  __shared__ bf16t Pp[8][16 * 32];

  const int tid = threadIdx.x, lane = tid & 31, wave = tid >> 5;
  int bid = blockIdx.x;
  int qblk = bid % nQB; bid /= nQB;
  int head = bid % NHEAD;
  int win  = bid / NHEAD;

  if (wave == 0) {   // K,V tiles via Tensor Data Mover (Nk x 32 bf16, row stride 576)
    unsigned long long base =
        (unsigned long long)(QKV + (size_t)(win * Nwin + kb) * 576 + head * HDIM);
    tdm_load_2d(lds_off32(&Ks[0]), base + 192ull * 2ull, (unsigned)Nk, 576u);
    tdm_load_2d(lds_off32(&Vs[0]), base + 384ull * 2ull, (unsigned)Nk, 576u);
    __builtin_amdgcn_s_wait_tensorcnt(0);
  }
  __syncthreads();

  const int qrow0 = qblk * 128 + wave * 16;
  if (qrow0 >= Nq) return;                             // wave-uniform
  const int hi8 = (lane >> 4) << 3;
  const int n16 = lane & 15;

  v16bf qa;                                            // Q A-fragment (16x32)
  {
    int qp = qb + qrow0 + n16;
    size_t base = (size_t)(win * Nwin + qp) * 576 + head * HDIM;
    for (int e = 0; e < 16; e++) qa[e] = QKV[base + kmap16(e, lane)];
  }

  v8f acc0 = zero8(), acc1 = zero8();
  float m[8], l[8];
  for (int r = 0; r < 8; r++) { m[r] = -1e30f; l[r] = 0.f; }

  for (int kt = 0; kt < (Nk >> 5); kt++) {
    v16bf bk0, bk1;                                    // K^T B-frags (dims x keys)
    for (int e = 0; e < 16; e++) {
      int km = kmap16(e, lane);
      bk0[e] = Ks[(kt * 32 + n16) * 32 + km];
      bk1[e] = Ks[(kt * 32 + 16 + n16) * 32 + km];
    }
    v8f s0 = wmma_bf16(qa, bk0, zero8());
    v8f s1 = wmma_bf16(qa, bk1, zero8());

    for (int r = 0; r < 8; r++) {                      // fp32 softmax (online)
      int rl = r + hi8;
      int ql = qrow0 + rl;
      int c0 = kt * 32 + n16, c1 = c0 + 16;
      float a0 = s0[r] * scale, a1 = s1[r] * scale;
      if (use_rpb) {
        a0 += RPB[rel_idx(qb + ql, kb + c0, ws0, ws1, ws2) * NHEAD + head];
        a1 += RPB[rel_idx(qb + ql, kb + c1, ws0, ws1, ws2) * NHEAD + head];
      }
      if (use_mask) {
        int lq = g1_label(win, ql);
        if (lq != g1_label(win, c0)) a0 -= 100.f;
        if (lq != g1_label(win, c1)) a1 -= 100.f;
      }
      float tm = fmaxf(a0, a1);
      for (int o = 1; o < 16; o <<= 1) tm = fmaxf(tm, __shfl_xor(tm, o));
      float mn = fmaxf(m[r], tm);
      float p0 = __expf(a0 - mn), p1 = __expf(a1 - mn);
      float ps = p0 + p1;
      for (int o = 1; o < 16; o <<= 1) ps += __shfl_xor(ps, o);
      float corr = __expf(m[r] - mn);
      l[r] = l[r] * corr + ps;
      m[r] = mn;
      acc0[r] *= corr; acc1[r] *= corr;
      Pp[wave][rl * 32 + n16]      = (bf16t)p0;        // bounce P -> A layout
      Pp[wave][rl * 32 + 16 + n16] = (bf16t)p1;
    }
    v16bf pa, bv0, bv1;
    for (int e = 0; e < 16; e++) {
      int km = kmap16(e, lane);
      pa[e]  = Pp[wave][n16 * 32 + km];
      bv0[e] = Vs[(kt * 32 + km) * 32 + n16];
      bv1[e] = Vs[(kt * 32 + km) * 32 + 16 + n16];
    }
    acc0 = wmma_bf16(pa, bv0, acc0);
    acc1 = wmma_bf16(pa, bv1, acc1);
  }

  for (int r = 0; r < 8; r++) {
    int ql = qrow0 + r + hi8;
    float inv = 1.f / l[r];
    int orow = ql + row_off;
    size_t ob = (size_t)(win * Nwin + orow) * Cout + col_off + head * HDIM;
    Out[ob + n16]      = (bf16t)(acc0[r] * inv);
    Out[ob + 16 + n16] = (bf16t)(acc1[r] * inv);
  }
}

// ---------------------------------------------------------------- host orchestration

struct WsBufs {
  float *Xcur, *Xsave, *Proj, *H1, *H2, *pos;
  bf16t *Yw, *Ywp, *QKV, *QKVm, *AttnB, *Hb, *Wb;
};

static void launch_gemm(const bf16t* A, const float* Wsrc, bf16t* Wscr,
                        const float* bias, float* Of, bf16t* Ob,
                        int M, int N, int K, hipStream_t stream) {
  int wn = K * N;
  to_bf16_kernel<<<(wn + 255) / 256, 256, 0, stream>>>(Wsrc, Wscr, wn);
  gemm_bias_kernel<<<dim3(N / 64, M / 128), 256, 0, stream>>>(A, Wscr, bias, Of, Ob, M, N, K);
}

static void launch_attn(const bf16t* QKV, const float* rpb, bf16t* Out,
                        int nW, int Nwin, int Nq, int Nk, int qb, int kb, int row_off,
                        int Cout, int col_off, int use_rpb, int use_mask,
                        int ws0, hipStream_t stream) {
  int nQB = (Nq + 127) / 128;
  attn_kernel<<<dim3(nW * NHEAD * nQB), 256, 0, stream>>>(
      QKV, rpb, Out, nW, Nwin, Nq, Nk, qb, kb, row_off, Cout, col_off,
      use_rpb, use_mask, ws0, 8, 8, 0.17677669529663689f, nQB);
}

static void run_block(const float* const* P, bool mut, int mode, bool shifted,
                      WsBufs& B, hipStream_t stream) {
  const int NE = TKN * CDIM, GE = (NE + 255) / 256;
  int s0 = shifted ? 1 : 0, s1 = shifted ? 4 : 0, s2 = shifted ? 4 : 0;
  // LN1 + window partition (+pos for mutual branch)
  ln_window_kernel<<<TKN / 8, 256, 0, stream>>>(B.Xcur, P[0], P[1], B.Yw,
                                                mut ? B.Ywp : nullptr, B.pos, mode, s0, s1, s2);
  launch_gemm(B.Yw, P[2], B.Wb, P[3], nullptr, B.QKV, TKN, 576, 192, stream);
  if (mut) launch_gemm(B.Ywp, P[15], B.Wb, P[16], nullptr, B.QKVm, TKN, 576, 192, stream);

  int nW   = (mode == 0) ? 192 : 64;
  int Nwin = (mode == 0) ? 128 : 384;
  int ws0  = (mode == 0) ? 2 : 6;
  int Cout = mut ? 384 : 192;
  int coff = mut ? 192 : 0;
  launch_attn(B.QKV, P[4], B.AttnB, nW, Nwin, Nwin, Nwin, 0, 0, 0,
              Cout, coff, 1, shifted ? 1 : 0, ws0, stream);
  if (mut) {   // x1: queries 64..127 over keys 0..63 -> rows 0..63; x2 mirrored
    launch_attn(B.QKVm, nullptr, B.AttnB, nW, Nwin, 64, 64, 64, 0, 0,
                Cout, 0, 0, shifted ? 1 : 0, ws0, stream);
    launch_attn(B.QKVm, nullptr, B.AttnB, nW, Nwin, 64, 64, 0, 64, 64,
                Cout, 0, 0, shifted ? 1 : 0, ws0, stream);
  }
  launch_gemm(B.AttnB, P[5], B.Wb, P[6], B.Proj, nullptr, TKN, 192, Cout, stream);
  unwindow_add_kernel<<<GE, 256, 0, stream>>>(B.Xcur, B.Proj, mode, s0, s1, s2);

  // GEGLU MLP
  ln_window_kernel<<<TKN / 8, 256, 0, stream>>>(B.Xcur, P[7], P[8], B.Yw, nullptr, B.pos, 2, 0, 0, 0);
  launch_gemm(B.Yw, P[9],  B.Wb, P[10], B.H1, nullptr, TKN, 384, 192, stream);
  launch_gemm(B.Yw, P[11], B.Wb, P[12], B.H2, nullptr, TKN, 384, 192, stream);
  geglu_kernel<<<(TKN * 384 + 255) / 256, 256, 0, stream>>>(B.H1, B.H2, B.Hb, TKN * 384);
  launch_gemm(B.Hb, P[13], B.Wb, P[14], B.Proj, nullptr, TKN, 192, 384, stream);
  add_kernel<<<GE, 256, 0, stream>>>(B.Xcur, B.Proj, NE);
}

extern "C" void kernel_launch(void* const* d_in, const int* in_sizes, int n_in,
                              void* d_out, int out_size, void* d_ws, size_t ws_size,
                              hipStream_t stream) {
  (void)in_sizes; (void)n_in; (void)out_size; (void)ws_size;
  const float* x = (const float*)d_in[0];
  const float* g1p[3][17];
  int idx = 1;
  for (int b = 0; b < 3; b++)
    for (int j = 0; j < 17; j++) g1p[b][j] = (const float*)d_in[idx++];
  const float* g2p[17] = {};
  for (int j = 0; j < 15; j++) g2p[j] = (const float*)d_in[idx++];
  const float* lin1w = (const float*)d_in[idx++];
  const float* lin1b = (const float*)d_in[idx++];
  const float* lin2w = (const float*)d_in[idx++];
  const float* lin2b = (const float*)d_in[idx++];

  char* wsb = (char*)d_ws;
  size_t off = 0;
  auto alloc = [&](size_t bytes) -> void* {
    off = (off + 255) & ~(size_t)255;
    void* p = wsb + off; off += bytes; return p;
  };
  WsBufs B;
  B.Xcur  = (float*)alloc((size_t)TKN * CDIM * 4);
  B.Xsave = (float*)alloc((size_t)TKN * CDIM * 4);
  B.Proj  = (float*)alloc((size_t)TKN * CDIM * 4);
  B.H1    = (float*)alloc((size_t)TKN * 384 * 4);
  B.H2    = (float*)alloc((size_t)TKN * 384 * 4);
  B.pos   = (float*)alloc((size_t)128 * CDIM * 4);
  B.Yw    = (bf16t*)alloc((size_t)TKN * CDIM * 2);
  B.Ywp   = (bf16t*)alloc((size_t)TKN * CDIM * 2);
  B.QKV   = (bf16t*)alloc((size_t)TKN * 576 * 2);
  B.QKVm  = (bf16t*)alloc((size_t)TKN * 576 * 2);
  B.AttnB = (bf16t*)alloc((size_t)TKN * 384 * 2);
  B.Hb    = (bf16t*)alloc((size_t)TKN * 384 * 2);
  B.Wb    = (bf16t*)alloc((size_t)192 * 576 * 2);

  const int NE = TKN * CDIM, GE = (NE + 255) / 256;
  transpose_in_kernel<<<GE, 256, 0, stream>>>(x, B.Xcur, NE);
  pos_kernel<<<(128 * CDIM + 255) / 256, 256, 0, stream>>>(B.pos);

  // ---- group 1 (3 mutual-attention blocks, ws (2,8,8), block 1 shifted)
  copy_kernel<<<GE, 256, 0, stream>>>(B.Xsave, B.Xcur, NE);
  for (int b = 0; b < 3; b++) run_block(g1p[b], true, 0, b == 1, B, stream);
  to_bf16_kernel<<<GE, 256, 0, stream>>>(B.Xcur, B.Yw, NE);
  launch_gemm(B.Yw, lin1w, B.Wb, lin1b, B.Proj, nullptr, TKN, 192, 192, stream);
  add2_kernel<<<GE, 256, 0, stream>>>(B.Xcur, B.Xsave, B.Proj, NE);

  // ---- group 2 (1 block, ws (6,8,8), no shift, no mutual)
  copy_kernel<<<GE, 256, 0, stream>>>(B.Xsave, B.Xcur, NE);
  run_block(g2p, false, 1, false, B, stream);
  to_bf16_kernel<<<GE, 256, 0, stream>>>(B.Xcur, B.Yw, NE);
  launch_gemm(B.Yw, lin2w, B.Wb, lin2b, B.Proj, nullptr, TKN, 192, 192, stream);
  add2_kernel<<<GE, 256, 0, stream>>>(B.Xcur, B.Xsave, B.Proj, NE);

  transpose_out_kernel<<<GE, 256, 0, stream>>>(B.Xcur, (float*)d_out, NE);
}
